// Attention_33122787787003
// MI455X (gfx1250) — matrix operand order repaired and checked by reading, not compiled, not run on hardware
//
#include <hip/hip_runtime.h>
#include <hip/hip_bf16.h>
#include <math.h>

// ---------------------------------------------------------------------------
// Problem constants (match reference)
// ---------------------------------------------------------------------------
#define BATCH 32
#define SEQ   2048
#define DIM   1024
#define HID   1024
#define MROWS (BATCH * SEQ)          // 65536 GEMM rows

// Scores-kernel tiling
#define TILE_M   128                 // rows per block (8 waves x 16 rows)
#define BLK      256                 // threads per block (8 wave32)
#define NCHUNK   64                  // columns (h) processed per N-chunk
#define NT       4                   // 16-wide tiles per N-chunk
#define KPANEL   128                 // K depth staged per W panel
#define NPANEL   (DIM / KPANEL)      // 8 panels per chunk
#define NPTOT    ((HID / NCHUNK) * NPANEL)   // 128 linearized panels
#define A_STRIDE 1032                // bf16 elements per LDS A row (pad 8)
#define W_STRIDE 136                 // bf16 elements per LDS W row (pad 8)

#define A_LDS_BYTES   (TILE_M * A_STRIDE * 2)            // 264192
#define W_PANEL_ELEMS (NCHUNK * W_STRIDE)                // 8704 bf16
#define W_LDS_BYTES   (2 * W_PANEL_ELEMS * 2)            // 34816 (double buf)
#define SCORES_LDS    (A_LDS_BYTES + W_LDS_BYTES)        // 299008 <= 320KB

// Async LDS-copy path (gfx1250 GLOBAL_LOAD_ASYNC_TO_LDS_B128), guarded.
#if __has_builtin(__builtin_amdgcn_global_load_async_to_lds_b128) && \
    __has_builtin(__builtin_amdgcn_s_wait_asynccnt)
#define USE_ASYNC_LDS 1
#else
#define USE_ASYNC_LDS 0
#endif

// Hardware tanh (CDNA5 V_TANH_F32 trans op), guarded.
#if __has_builtin(__builtin_amdgcn_tanhf)
#define FAST_TANH(x) __builtin_amdgcn_tanhf(x)
#elif __has_builtin(__builtin_amdgcn_tanh_f32)
#define FAST_TANH(x) __builtin_amdgcn_tanh_f32(x)
#else
#define FAST_TANH(x) tanhf(x)
#endif

// ---------------------------------------------------------------------------
// Vector / WMMA fragment types (CDNA5 gfx1250, wave32)
// ---------------------------------------------------------------------------
typedef __attribute__((ext_vector_type(16))) __bf16 v16bf;
typedef __attribute__((ext_vector_type(8)))  float  v8f;
typedef __attribute__((ext_vector_type(4)))  float  v4f;
typedef __attribute__((ext_vector_type(4)))  int    v4i;

union FragBF {          // 16 bf16 = 32B = two b128 chunks
    uint4 u[2];
    v16bf v;
};

struct FragSet {        // one k-step worth of operands: A + NT B tiles
    FragBF a;
    FragBF b[NT];
};

// f32 -> bf16 (round to nearest even), packed pair helper
__device__ __forceinline__ unsigned int pack_bf16(float lo, float hi) {
    unsigned int ulo = __float_as_uint(lo);
    unsigned int uhi = __float_as_uint(hi);
    ulo = (ulo + 0x7FFFu + ((ulo >> 16) & 1u)) >> 16;
    uhi = (uhi + 0x7FFFu + ((uhi >> 16) & 1u)) >> 16;
    return (uhi << 16) | (ulo & 0xFFFFu);
}

// ---------------------------------------------------------------------------
// Kernel 1: inp = input @ W_in.T + b_in   ([32,1024], tiny)
// ---------------------------------------------------------------------------
__global__ __launch_bounds__(256) void input_linear_kernel(
    const float* __restrict__ input, const float* __restrict__ W_in,
    const float* __restrict__ b_in, float* __restrict__ inp_act)
{
    __shared__ float xrow[DIM];
    const int b     = blockIdx.x >> 2;
    const int hbase = (blockIdx.x & 3) * 256;
    const int t     = threadIdx.x;

    for (int i = t; i < DIM; i += 256)
        xrow[i] = input[(size_t)b * DIM + i];
    __syncthreads();

    const int h = hbase + t;
    const float4* wr = (const float4*)(W_in + (size_t)h * DIM);
    const float4* xr = (const float4*)xrow;
    float acc = 0.0f;
#pragma unroll 8
    for (int d = 0; d < DIM / 4; ++d) {
        float4 w = wr[d];
        float4 x = xr[d];
        acc += w.x * x.x + w.y * x.y + w.z * x.z + w.w * x.w;
    }
    inp_act[(size_t)b * HID + h] = acc + b_in[h];
}

// ---------------------------------------------------------------------------
// Kernel 1b: one-shot W_ctx f32 -> bf16 (row-major [H,D], 2MB, L2-resident)
// ---------------------------------------------------------------------------
__global__ __launch_bounds__(256) void wconvert_kernel(
    const float* __restrict__ W, unsigned short* __restrict__ Wb)
{
    const size_t idx = ((size_t)blockIdx.x * 256 + threadIdx.x) * 8;
    const float4* s = (const float4*)(W + idx);
    float4 a = s[0], c = s[1];
    uint4 p;
    p.x = pack_bf16(a.x, a.y);
    p.y = pack_bf16(a.z, a.w);
    p.z = pack_bf16(c.x, c.y);
    p.w = pack_bf16(c.z, c.w);
    *(uint4*)(Wb + idx) = p;
}

// ---------------------------------------------------------------------------
// W panel staging helpers (64 cols x 128 K of bf16, padded stride W_STRIDE)
// ---------------------------------------------------------------------------
#if USE_ASYNC_LDS
__device__ __forceinline__ void stage_panel_async(
    const unsigned short* __restrict__ wsrc,   // bf16 [64][DIM] panel origin
    unsigned short* __restrict__ wbuf,         // LDS dest
    int t)
{
#pragma unroll
    for (int i = 0; i < 4; ++i) {              // 1024 x 16B chunks / 256 thr
        int idx = t + i * BLK;
        int row = idx >> 4;                    // 16 chunks of 8 bf16 per row
        int c8  = idx & 15;
        v4i* g = (v4i*)(wsrc + (size_t)row * DIM + c8 * 8);
        v4i* l = (v4i*)(wbuf + row * W_STRIDE + c8 * 8);
        __builtin_amdgcn_global_load_async_to_lds_b128(g, l, 0, 0);
    }
}
#else
__device__ __forceinline__ void load_panel_regs(
    const unsigned short* __restrict__ wsrc, int t, uint4* st)
{
#pragma unroll
    for (int i = 0; i < 4; ++i) {
        int idx = t + i * BLK;
        int row = idx >> 4;
        int c8  = idx & 15;
        st[i] = *(const uint4*)(wsrc + (size_t)row * DIM + c8 * 8);
    }
}
__device__ __forceinline__ void store_panel_regs(
    unsigned short* __restrict__ wbuf, int t, const uint4* st)
{
#pragma unroll
    for (int i = 0; i < 4; ++i) {
        int idx = t + i * BLK;
        int row = idx >> 4;
        int c8  = idx & 15;
        *(uint4*)(wbuf + row * W_STRIDE + c8 * 8) = st[i];
    }
}
#endif

// Load one k-step's A + 4 B fragments from LDS (all b128, aligned)
__device__ __forceinline__ void load_fragset(
    const unsigned short* __restrict__ arow0,
    const unsigned short* __restrict__ wrow0,
    int ks, FragSet& f)
{
    const unsigned short* an = arow0 + ks * 32;
    f.a.u[0] = *(const uint4*)(an);
    f.a.u[1] = *(const uint4*)(an + 16);
#pragma unroll
    for (int tt = 0; tt < NT; ++tt) {
        const unsigned short* w = wrow0 + tt * 16 * W_STRIDE + ks * 32;
        f.b[tt].u[0] = *(const uint4*)(w);
        f.b[tt].u[1] = *(const uint4*)(w + 16);
    }
}

// ---------------------------------------------------------------------------
// Kernel 2: fused ctx-GEMM (bf16 WMMA) + tanh + V-dot + mask -> att[B*S]
// grid = MROWS/TILE_M = 512 blocks, 256 threads (8 waves)
// ---------------------------------------------------------------------------
__global__ __launch_bounds__(BLK, 1) void attn_scores_kernel(
    const float* __restrict__ ctx,            // [B*S, D] f32
    const unsigned short* __restrict__ Wbf,   // [H, D]  bf16 (pre-converted)
    const float* __restrict__ b_ctx,          // [H]
    const float* __restrict__ Vvec,           // [H]
    const float* __restrict__ inp_act,        // [B, H]
    const int*   __restrict__ mask,           // [B*S]
    float* __restrict__ att_out)              // [B*S]
{
    extern __shared__ char smem[];
    unsigned short* As = (unsigned short*)smem;
    unsigned short* Ws = (unsigned short*)(smem + A_LDS_BYTES);

    const int t      = threadIdx.x;
    const int wave   = t >> 5;
    const int lane   = t & 31;
    const int n_lane = lane & 15;   // column within 16-wide tile
    const int khalf  = lane >> 4;   // K-half select (ISA 16-bit frag layout)
    const int mbase  = blockIdx.x * TILE_M;
    const int b      = mbase / SEQ; // whole block sits inside one batch
    const int row0   = wave * 16;   // this wave's M sub-tile (local rows)

    // ---- Stage A: 128 rows x 1024 f32 -> bf16 LDS (streamed, non-temporal)
    {
        const v4f* src = (const v4f*)(ctx + (size_t)mbase * DIM);
#pragma unroll 2
        for (int i = 0; i < (TILE_M * DIM / 4) / BLK; ++i) {   // 128 iters
            int idx = t + i * BLK;
            int row = idx >> 8;
            int c4  = idx & 255;
            v4f v = __builtin_nontemporal_load(
                src + (size_t)row * (DIM / 4) + c4);
            uint2 p;
            p.x = pack_bf16(v.x, v.y);
            p.y = pack_bf16(v.z, v.w);
            *(uint2*)(As + row * A_STRIDE + c4 * 4) = p;
        }
    }

    float rowacc[8];
#pragma unroll
    for (int r = 0; r < 8; ++r) rowacc[r] = 0.0f;

    v8f acc[NT];
#if !USE_ASYNC_LDS
    uint4 streg[4];
#endif

    // ---- prologue: stage linear panel 0 ----
#if USE_ASYNC_LDS
    stage_panel_async(Wbf, Ws, t);
    __builtin_amdgcn_s_wait_asynccnt(0);
#else
    load_panel_regs(Wbf, t, streg);
    store_panel_regs(Ws, t, streg);
#endif
    __syncthreads();

    // ---- linearized panel loop: p = nc*NPANEL + kp over all of H x D ----
    for (int p = 0; p < NPTOT; ++p) {
        const int kp = p & (NPANEL - 1);
        const int nc = p >> 3;                       // NPANEL == 8
        unsigned short* Wbuf = Ws + (p & 1) * W_PANEL_ELEMS;

        if (kp == 0) {
#pragma unroll
            for (int tt = 0; tt < NT; ++tt)
#pragma unroll
                for (int r = 0; r < 8; ++r) acc[tt][r] = 0.0f;
        }

        // kick off next panel copy (overlaps WMMAs, and at chunk boundaries
        // overlaps the tanh epilogue as well)
        if (p + 1 < NPTOT) {
            const int pn = p + 1;
            const unsigned short* nsrc =
                Wbf + (size_t)((pn >> 3) * NCHUNK) * DIM +
                (pn & (NPANEL - 1)) * KPANEL;
#if USE_ASYNC_LDS
            stage_panel_async(nsrc, Ws + (pn & 1) * W_PANEL_ELEMS, t);
#else
            load_panel_regs(nsrc, t, streg);
#endif
        }

        // ---- 4 WMMA K-steps, 2-deep fragment pipeline in registers ----
        const unsigned short* arow0 =
            As + (row0 + n_lane) * A_STRIDE + kp * KPANEL + khalf * 8;
        const unsigned short* wrow0 = Wbuf + n_lane * W_STRIDE + khalf * 8;

        FragSet fr[2];
        load_fragset(arow0, wrow0, 0, fr[0]);
#pragma unroll
        for (int ks = 0; ks < KPANEL / 32; ++ks) {
            if (ks + 1 < KPANEL / 32)
                load_fragset(arow0, wrow0, ks + 1, fr[(ks + 1) & 1]);
            FragSet& cur = fr[ks & 1];
#pragma unroll
            for (int tt = 0; tt < NT; ++tt)
                acc[tt] = __builtin_amdgcn_wmma_f32_16x16x32_bf16(
                    false, cur.a.v, false, cur.b[tt].v,
                    (short)0, acc[tt], false, false);
        }

        // ---- chunk epilogue: rowacc += V[h]*tanh(acc + b_ctx[h] + inp[b,h])
        if (kp == NPANEL - 1) {
            const int hbase = nc * NCHUNK;
#pragma unroll
            for (int tt = 0; tt < NT; ++tt) {
                const int h = hbase + tt * 16 + n_lane;
                const float cadd = b_ctx[h] + inp_act[(size_t)b * HID + h];
                const float vv   = Vvec[h];
#pragma unroll
                for (int r = 0; r < 8; ++r)
                    rowacc[r] += vv * FAST_TANH(acc[tt][r] + cadd);
            }
        }

        // drain the next-panel copy, release buffers
#if USE_ASYNC_LDS
        if (p + 1 < NPTOT)
            __builtin_amdgcn_s_wait_asynccnt(0);
#else
        if (p + 1 < NPTOT)
            store_panel_regs(Ws + ((p + 1) & 1) * W_PANEL_ELEMS, t, streg);
#endif
        __syncthreads();
    }

    // ---- reduce 16 lanes per half -> one att scalar per row, apply mask ----
#pragma unroll
    for (int r = 0; r < 8; ++r) {
        float s = rowacc[r];
        s += __shfl_xor(s, 1, 32);
        s += __shfl_xor(s, 2, 32);
        s += __shfl_xor(s, 4, 32);
        s += __shfl_xor(s, 8, 32);
        if (n_lane == 0) {
            const int m = mbase + row0 + khalf * 8 + r;
            att_out[m] = mask[m] ? -INFINITY : s;
        }
    }
}

// ---------------------------------------------------------------------------
// Kernel 3: softmax over S per batch row. grid=32, block=256.
// ---------------------------------------------------------------------------
__global__ __launch_bounds__(256) void softmax_kernel(
    const float* __restrict__ att, float* __restrict__ alpha)
{
    __shared__ float red[256];
    const int b = blockIdx.x;
    const int t = threadIdx.x;
    const float* row = att + (size_t)b * SEQ;

    float vals[SEQ / 256];
    float lmax = -INFINITY;
#pragma unroll
    for (int i = 0; i < SEQ / 256; ++i) {
        vals[i] = row[t + i * 256];
        lmax = fmaxf(lmax, vals[i]);
    }
    red[t] = lmax;
    __syncthreads();
    for (int off = 128; off > 0; off >>= 1) {
        if (t < off) red[t] = fmaxf(red[t], red[t + off]);
        __syncthreads();
    }
    const float m = red[0];
    __syncthreads();

    float lsum = 0.0f;
#pragma unroll
    for (int i = 0; i < SEQ / 256; ++i) {
        float e = __expf(vals[i] - m);
        vals[i] = e;
        lsum += e;
    }
    red[t] = lsum;
    __syncthreads();
    for (int off = 128; off > 0; off >>= 1) {
        if (t < off) red[t] += red[t + off];
        __syncthreads();
    }
    const float inv = 1.0f / red[0];
#pragma unroll
    for (int i = 0; i < SEQ / 256; ++i)
        alpha[(size_t)b * SEQ + t + i * 256] = vals[i] * inv;
}

// ---------------------------------------------------------------------------
// Launcher
// ---------------------------------------------------------------------------
extern "C" void kernel_launch(void* const* d_in, const int* in_sizes, int n_in,
                              void* d_out, int out_size, void* d_ws,
                              size_t ws_size, hipStream_t stream)
{
    const float* input   = (const float*)d_in[0];  // [B, D]
    const float* context = (const float*)d_in[1];  // [B, S, D]
    const int*   mask    = (const int*)  d_in[2];  // [B, S]
    const float* W_in    = (const float*)d_in[3];  // [H, D]
    const float* b_in    = (const float*)d_in[4];  // [H]
    const float* W_ctx   = (const float*)d_in[5];  // [H, D]
    const float* b_ctx   = (const float*)d_in[6];  // [H]
    const float* V       = (const float*)d_in[7];  // [H]

    float* alpha = (float*)d_out;                  // [B*S]
    float* att   = (float*)d_out + MROWS;          // [B*S]

    float*          inp_act = (float*)d_ws;                       // 128 KB
    unsigned short* Wbf     = (unsigned short*)((char*)d_ws + BATCH * HID *
                                                sizeof(float));   // 2 MB bf16

    static const size_t scores_lds = SCORES_LDS;
    (void)hipFuncSetAttribute((const void*)attn_scores_kernel,
                              hipFuncAttributeMaxDynamicSharedMemorySize,
                              (int)scores_lds);

    input_linear_kernel<<<BATCH * 4, 256, 0, stream>>>(input, W_in, b_in,
                                                       inp_act);
    wconvert_kernel<<<(HID * DIM / 8) / 256, 256, 0, stream>>>(W_ctx, Wbf);
    attn_scores_kernel<<<MROWS / TILE_M, BLK, scores_lds, stream>>>(
        context, Wbf, b_ctx, V, inp_act, mask, att);
    softmax_kernel<<<BATCH, 256, 0, stream>>>(att, alpha);
}
